// IGAE_decoder_67070209294348
// MI455X (gfx1250) — matrix operand order repaired and checked by reading, not compiled, not run on hardware
//
#include <hip/hip_runtime.h>

typedef __bf16 bf16;
typedef __attribute__((ext_vector_type(16))) __bf16 v16bf;
typedef __attribute__((ext_vector_type(8)))  __bf16 v8bf;
typedef __attribute__((ext_vector_type(8)))  float  v8f;

__device__ __forceinline__ unsigned short f32_to_bf16_bits(float f) {
    union { float f; unsigned u; } in; in.f = f;
    unsigned u = in.u;
    u += 0x7FFFu + ((u >> 16) & 1u);          // round-to-nearest-even
    return (unsigned short)(u >> 16);
}

// ---- cast f32[rows,K] -> bf16[rows,Kp], zero-padding K..Kp (K%4==0, Kp%8==0) ----
__global__ __launch_bounds__(256)
void cast_pad_bf16(const float* __restrict__ s, bf16* __restrict__ d,
                   int rows, int K, int Kp) {
    long long gid = (long long)blockIdx.x * 256 + threadIdx.x;
    int cpr = Kp >> 2;
    if (gid >= (long long)rows * cpr) return;
    int r  = (int)(gid / cpr);
    int c4 = (int)(gid % cpr) * 4;
    uint2 p;
    if (c4 < K) {
        float4 v = *(const float4*)(s + (long long)r * K + c4);
        p.x = (unsigned)f32_to_bf16_bits(v.x) | ((unsigned)f32_to_bf16_bits(v.y) << 16);
        p.y = (unsigned)f32_to_bf16_bits(v.z) | ((unsigned)f32_to_bf16_bits(v.w) << 16);
    } else {
        p.x = 0; p.y = 0;
    }
    *(uint2*)(d + (long long)r * Kp + c4) = p;
}

// ---- cast + transpose + pad: W[K,N] f32 -> out[N,Kp] bf16 ----
__global__ __launch_bounds__(256)
void cast_transpose_bf16(const float* __restrict__ W, bf16* __restrict__ out,
                         int K, int N, int Kp) {
    long long gid = (long long)blockIdx.x * 256 + threadIdx.x;
    if (gid >= (long long)N * Kp) return;
    int n = (int)(gid / Kp);
    int k = (int)(gid % Kp);
    unsigned short b = (k < K) ? f32_to_bf16_bits(W[(long long)k * N + n]) : (unsigned short)0;
    ((unsigned short*)out)[gid] = b;
}

#define BM 128
#define BN 256
#define BK 32

// C[M,N] = act(A @ B^T): A[M,Kp] bf16, B[N,Kp] bf16 (K pre-padded with zeros, Kp%32==0).
// Async (ASYNCcnt) global->LDS staging, double-buffered LDS, bf16 WMMA, f32 accum.
// ACT: 0 none, 1 leaky(0.2), 2 sigmoid (+ nontemporal store).
template<int ACT>
__global__ __launch_bounds__(256)
void gemm_nt_bf16(const bf16* __restrict__ A, const bf16* __restrict__ B,
                  float* __restrict__ C, int M, int N, int Kp) {
    __shared__ __attribute__((aligned(128))) bf16 lds_a[2][BM][BK];  // 2 x 8KB
    __shared__ __attribute__((aligned(128))) bf16 lds_b[2][BN][BK];  // 2 x 16KB

    const int tid  = threadIdx.x;
    const int lane = tid & 31;
    const int wave = tid >> 5;
    const int wm   = wave & 1;     // 2 wave-rows -> 64 M each
    const int wn   = wave >> 1;    // 4 wave-cols -> 64 N each
    const int m0   = blockIdx.y * BM;
    const int n0   = blockIdx.x * BN;
    const int lr   = lane & 15;
    const int lh   = lane >> 4;

    // Per-thread staging coordinates (16B chunks along K)
    const int ar  = tid >> 2;             // A rows: 2 iters cover 128
    const int aj8 = (tid & 3) * 8;
    int gmA0 = m0 + ar;       gmA0 = gmA0 < M ? gmA0 : M - 1;
    int gmA1 = m0 + ar + 64;  gmA1 = gmA1 < M ? gmA1 : M - 1;
    long long aoff0 = (long long)gmA0 * Kp + aj8;
    long long aoff1 = (long long)gmA1 * Kp + aj8;

    // Issue one stage of async copies: 2 A-chunks + 4 B-chunks per thread (24KB/block)
    auto stage_async = [&](int k0, int buf) {
        {
            unsigned l0 = (unsigned)(size_t)&lds_a[buf][ar][aj8];
            unsigned l1 = (unsigned)(size_t)&lds_a[buf][ar + 64][aj8];
            int g0 = (int)((aoff0 + k0) * 2);
            int g1 = (int)((aoff1 + k0) * 2);
            asm volatile("global_load_async_to_lds_b128 %0, %1, %2"
                         :: "v"(l0), "v"(g0), "s"(A) : "memory");
            asm volatile("global_load_async_to_lds_b128 %0, %1, %2"
                         :: "v"(l1), "v"(g1), "s"(A) : "memory");
        }
#pragma unroll
        for (int t = 0; t < 4; ++t) {
            int c = (tid >> 2) + t * 64;
            int gn = n0 + c; gn = gn < N ? gn : N - 1;
            unsigned l = (unsigned)(size_t)&lds_b[buf][c][aj8];
            int g = (int)(((long long)gn * Kp + k0 + aj8) * 2);
            asm volatile("global_load_async_to_lds_b128 %0, %1, %2"
                         :: "v"(l), "v"(g), "s"(B) : "memory");
        }
    };

    v8f acc[4][4];
    const v8f vzero = {0.f,0.f,0.f,0.f,0.f,0.f,0.f,0.f};
#pragma unroll
    for (int mi = 0; mi < 4; ++mi)
#pragma unroll
        for (int ni = 0; ni < 4; ++ni) acc[mi][ni] = vzero;

    stage_async(0, 0);                      // prologue prefetch
    int buf = 0;
    for (int k0 = 0; k0 < Kp; k0 += BK) {
        if (k0 + BK < Kp) {
            stage_async(k0 + BK, buf ^ 1);  // prefetch next stage under compute
            asm volatile("s_wait_asynccnt 6" ::: "memory");   // current stage's 6 done
        } else {
            asm volatile("s_wait_asynccnt 0" ::: "memory");
        }
        __syncthreads();                    // stage visible to all waves

        // B fragments: lane holds col=lr, K = lh*16 + j (contiguous 32B)
        v16bf bfrag[4];
#pragma unroll
        for (int ni = 0; ni < 4; ++ni) {
            int c = wn * 64 + ni * 16 + lr;
            bfrag[ni] = *(const v16bf*)&lds_b[buf][c][lh * 16];
        }
        // A fragments: lane holds row=lr, K = lh*16 + j (contiguous 32B)
#pragma unroll
        for (int mi = 0; mi < 4; ++mi) {
            int r = wm * 64 + mi * 16 + lr;
            v8bf lo = *(const v8bf*)&lds_a[buf][r][lh * 16];
            v8bf hi = *(const v8bf*)&lds_a[buf][r][lh * 16 + 8];
            v16bf afrag = __builtin_shufflevector(lo, hi,
                0, 1, 2, 3, 4, 5, 6, 7, 8, 9, 10, 11, 12, 13, 14, 15);
#pragma unroll
            for (int ni = 0; ni < 4; ++ni) {
                acc[mi][ni] = __builtin_amdgcn_wmma_f32_16x16x32_bf16(
                    false, afrag, false, bfrag[ni], (short)0, acc[mi][ni],
                    false, false);
            }
        }
        __syncthreads();                    // all waves done reading before overwrite
        buf ^= 1;
    }

    // ---- epilogue: C/D layout row = i + 8*lh, col = lr ----
#pragma unroll
    for (int mi = 0; mi < 4; ++mi) {
#pragma unroll
        for (int ni = 0; ni < 4; ++ni) {
            int gc = n0 + wn * 64 + ni * 16 + lr;
            if (gc >= N) continue;
#pragma unroll
            for (int i = 0; i < 8; ++i) {
                int gr = m0 + wm * 64 + mi * 16 + i + 8 * lh;
                if (gr < M) {
                    float x = acc[mi][ni][i];
                    if (ACT == 1) x = (x > 0.f) ? x : 0.2f * x;
                    else if (ACT == 2) x = 1.0f / (1.0f + __expf(-x));
                    float* p = &C[(long long)gr * N + gc];
                    if (ACT == 2) __builtin_nontemporal_store(x, p);  // 400MB stream
                    else *p = x;
                }
            }
        }
    }
}

// Y[row[e], :] += val[e] * X[col[e], :]   (COO scatter; Y pre-zeroed; L2-resident)
__global__ __launch_bounds__(256)
void spmm_atomic(const int* __restrict__ rows, const int* __restrict__ cols,
                 const float* __restrict__ vals, const float* __restrict__ X,
                 float* __restrict__ Y, int E, int F) {
    long long gid = (long long)blockIdx.x * blockDim.x + threadIdx.x;
    int chunks = (F + 3) >> 2;
    long long total = (long long)E * chunks;
    if (gid >= total) return;
    int e  = (int)(gid / chunks);
    int c0 = (int)(gid % chunks) * 4;
    int r  = rows[e];
    int cl = cols[e];
    float v = vals[e];
    const float* xr = X + (long long)cl * F;
    float*       yr = Y + (long long)r  * F;
#pragma unroll
    for (int j = 0; j < 4; ++j) {
        int c = c0 + j;
        if (c < F) atomicAdd(&yr[c], v * xr[c]);
    }
}

__global__ __launch_bounds__(256)
void zero_f32(float* __restrict__ p, long long n) {
    long long i = (long long)blockIdx.x * blockDim.x + threadIdx.x;
    if (i < n) p[i] = 0.f;
}

extern "C" void kernel_launch(void* const* d_in, const int* in_sizes, int n_in,
                              void* d_out, int out_size, void* d_ws, size_t ws_size,
                              hipStream_t stream) {
    const float* z_igae = (const float*)d_in[0];
    const float* W4     = (const float*)d_in[1];
    const float* W5     = (const float*)d_in[2];
    const float* W6     = (const float*)d_in[3];
    const float* evals  = (const float*)d_in[4];
    const int*   erow   = (const int*)d_in[5];
    const int*   ecol   = (const int*)d_in[6];

    const int NN = in_sizes[0] / 32;   // 10000 nodes
    const int E  = in_sizes[4];        // 320000 edges
    const int F0 = 32, F1 = 128, F2 = 256, F3 = 3000;
    const int F3p = 3008;              // F3 padded to 32 for async staging

    // Output tuple layout: z_hat, z_hat_adj, az_1, az_2, az_3, z1, z2
    float* out = (float*)d_out;
    long long off = 0;
    float* z_hat = out + off; off += (long long)NN * F3;
    float* z_adj = out + off; off += (long long)NN * NN;
    float* az1   = out + off; off += (long long)NN * F1;
    float* az2   = out + off; off += (long long)NN * F2;
    float* az3   = out + off; off += (long long)NN * F3;
    float* z1    = out + off; off += (long long)NN * F1;
    float* z2    = out + off;

    // Workspace: sup f32 (120MB) | bfA bf16 (NN*F3p = 60.2MB) | bfB bf16 (<=1.5MB)
    float* sup = (float*)d_ws;
    bf16* bfA = (bf16*)((char*)d_ws + (size_t)NN * F3 * sizeof(float));
    bf16* bfB = bfA + (long long)NN * F3p;

    dim3 blk(256);
    auto ggrid = [](int M, int N) { return dim3((unsigned)((N + BN - 1) / BN),
                                                (unsigned)((M + BM - 1) / BM)); };
    auto cast = [&](const float* s, bf16* d, int rows, int K, int Kp) {
        long long n = (long long)rows * (Kp >> 2);
        cast_pad_bf16<<<dim3((unsigned)((n + 255) / 256)), blk, 0, stream>>>(s, d, rows, K, Kp);
    };
    auto castT = [&](const float* W, bf16* d, int K, int N, int Kp) {
        long long n = (long long)N * Kp;
        cast_transpose_bf16<<<dim3((unsigned)((n + 255) / 256)), blk, 0, stream>>>(W, d, K, N, Kp);
    };
    auto run_spmm = [&](float* Y, const float* X, int F) {
        long long ny = (long long)NN * F;
        zero_f32<<<dim3((unsigned)((ny + 255) / 256)), blk, 0, stream>>>(Y, ny);
        long long tot = (long long)E * ((F + 3) >> 2);
        spmm_atomic<<<dim3((unsigned)((tot + 255) / 256)), blk, 0, stream>>>(
            erow, ecol, evals, X, Y, E, F);
    };

    // Layer 1: support = leaky(z_igae @ W4); z1 = A@support; az1 = A@z1
    cast(z_igae, bfA, NN, F0, F0);
    castT(W4, bfB, F0, F1, F0);
    gemm_nt_bf16<1><<<ggrid(NN, F1), blk, 0, stream>>>(bfA, bfB, sup, NN, F1, F0);
    run_spmm(z1,  sup, F1);
    run_spmm(az1, z1,  F1);
    // Layer 2
    cast(z1, bfA, NN, F1, F1);
    castT(W5, bfB, F1, F2, F1);
    gemm_nt_bf16<1><<<ggrid(NN, F2), blk, 0, stream>>>(bfA, bfB, sup, NN, F2, F1);
    run_spmm(z2,  sup, F2);
    run_spmm(az2, z2,  F2);
    // Layer 3
    cast(z2, bfA, NN, F2, F2);
    castT(W6, bfB, F2, F3, F2);
    gemm_nt_bf16<1><<<ggrid(NN, F3), blk, 0, stream>>>(bfA, bfB, sup, NN, F3, F2);
    run_spmm(z_hat, sup,   F3);
    run_spmm(az3,   z_hat, F3);
    // Dense reconstruction: sigmoid(z_hat @ z_hat^T) — 600 GFLOP bf16 WMMA, NT output stream
    cast(z_hat, bfA, NN, F3, F3p);
    gemm_nt_bf16<2><<<ggrid(NN, NN), blk, 0, stream>>>(bfA, bfA, z_adj, NN, NN, F3p);
}
